// VelPreVecPot2D_18021682774228
// MI455X (gfx1250) — matrix operand order repaired and checked by reading, not compiled, not run on hardware
//
#include <hip/hip_runtime.h>

typedef __attribute__((ext_vector_type(16))) _Float16 v16h;
typedef __attribute__((ext_vector_type(8)))  _Float16 v8h;
typedef __attribute__((ext_vector_type(2)))  _Float16 h2v;
typedef __attribute__((ext_vector_type(8)))  float    v8f;
typedef __attribute__((ext_vector_type(8)))  unsigned u32x8;
typedef __attribute__((ext_vector_type(4)))  unsigned u32x4;

#define NPTS 524288
#define NT   (NPTS / 16)
#define NW   2          // waves per block
#define TPW  2          // tiles per wave

#if __has_builtin(__builtin_amdgcn_wmma_f16_16x16x32_f16)
#define HAVE_WMMA_F16D 1
#else
#define HAVE_WMMA_F16D 0
#endif

// ---------------- helpers ----------------

static __device__ __forceinline__ v8f wmma16(v16h a, v16h b, v8f c) {
  // (neg_a, A, neg_b, B, c_mod, C, reuse_a, reuse_b)
  return __builtin_amdgcn_wmma_f32_16x16x32_f16(false, a, false, b, (short)0, c, false, false);
}

#if HAVE_WMMA_F16D
static __device__ __forceinline__ v8h wmma16h(v16h a, v16h b, v8h c) {
  return __builtin_amdgcn_wmma_f16_16x16x32_f16(false, a, false, b, (short)0, c, false, false);
}
#endif

static __device__ __forceinline__ unsigned pk2(float a, float b) {
  return __builtin_bit_cast(unsigned, __builtin_amdgcn_cvt_pkrtz(a, b));
}

static __device__ __forceinline__ float mytanh(float v) {
#if __has_builtin(__builtin_amdgcn_tanhf)
  return __builtin_amdgcn_tanhf(v); // native V_TANH_F32
#else
  v = fminf(9.0f, fmaxf(-9.0f, v));
  float e = __builtin_amdgcn_exp2f(v * 2.8853900817779268f); // 2*log2(e)
  return (e - 1.0f) * __builtin_amdgcn_rcpf(e + 1.0f);
#endif
}

static __device__ __forceinline__ v16h loadA(const _Float16* p) {
  return *(const v16h*)p;
}

static __device__ __forceinline__ v8f vzero() {
  v8f c;
#pragma unroll
  for (int r = 0; r < 8; ++r) c[r] = 0.0f;
  return c;
}

#if HAVE_WMMA_F16D
static __device__ __forceinline__ v8h hzero() {
  v8h c;
#pragma unroll
  for (int r = 0; r < 8; ++r) c[r] = (_Float16)0.0f;
  return c;
}
#endif

// Scheduling fence (no instructions): bounds the load-hoisting window so
// register pressure stays low.
static __device__ __forceinline__ void sched_fence() {
  asm volatile("" ::: "memory");
}

// Convert a 16x64 D-layout tile (4 tiles of f16 feature-pair dwords
// pk[j][q] = {feat 16j+2q(+lane-half*8), +1}) into the two 64x16 B-layout
// operands (K chunks 0..31 / 32..63). B dwords are this lane's own pk dwords;
// the SWAPX16 results are unused; no LDS memory traffic.
static __device__ __forceinline__ void buildB(const unsigned pk[4][4], bool hib, v16h* B) {
  unsigned sw[4][4];
#pragma unroll
  for (int j = 0; j < 4; ++j)
#pragma unroll
    for (int q = 0; q < 4; ++q)
      sw[j][q] = (unsigned)__builtin_amdgcn_ds_swizzle((int)pk[j][q], 0x401f); // SWAPX16
#pragma unroll
  for (int c = 0; c < 2; ++c) {
    u32x8 b;
#pragma unroll
    for (int v = 0; v < 8; ++v) {
      int q = v & 3, up = v >> 2;
      unsigned lov = up ? pk[2 * c + 1][q] : pk[2 * c][q];
      unsigned hiv = up ? pk[2 * c + 1][q] : pk[2 * c][q];
      b[v] = hib ? hiv : lov;
    }
    B[c] = __builtin_bit_cast(v16h, b);
  }
}

// ---------------- weight prep: build A-layout f16 operands in d_ws ----------------
// Regions (element units):
//  fwdA : 7*4*2*32*16 = 28672 halves  A = Ws[L]^T  (A[m][k] = Ws[L][k][m])
//  bwdA : 28672 halves                A = Ws[L]    (A[m][k] = Ws[L][m][k])
//  inA  : 4*32*16 = 2048 halves       A[m][k] = k<3 ? W_in[k][m] : (k==3 ? b_in[m] : 0)
//  outA : 2*32*16 = 1024 halves       A[m][k] = m<3 ? W_in[m][k] : 0
//  fwdBias : 7*4*32*8 = 7168 floats   per-lane C-layout splat of bs
//  gInit   : 2*4*32*8 = 2048 floats   per-lane C-layout splat of W_out[:,c]
__global__ void prep_kernel(const float* __restrict__ W_in, const float* __restrict__ b_in,
                            const float* __restrict__ Ws,  const float* __restrict__ bs,
                            const float* __restrict__ W_out,
                            _Float16* __restrict__ fwdA, _Float16* __restrict__ bwdA,
                            _Float16* __restrict__ inA,  _Float16* __restrict__ outA,
                            float* __restrict__ fwdBias, float* __restrict__ gInit) {
  const int total = 28672 + 28672 + 2048 + 1024 + 7168 + 2048; // 69632
  for (int i = blockIdx.x * blockDim.x + threadIdx.x; i < total; i += gridDim.x * blockDim.x) {
    int xx = i;
    if (xx < 28672) {
      int e = xx & 15, lane = (xx >> 4) & 31, kc = (xx >> 9) & 1, m = (xx >> 10) & 3, L = xx >> 12;
      int M = 16 * m + (lane & 15);
      int K = 32 * kc + 8 * (lane >> 4) + (e < 8 ? e : e + 8);
      fwdA[xx] = (_Float16)Ws[(L * 64 + K) * 64 + M];
      continue;
    }
    xx -= 28672;
    if (xx < 28672) {
      int e = xx & 15, lane = (xx >> 4) & 31, kc = (xx >> 9) & 1, m = (xx >> 10) & 3, L = xx >> 12;
      int M = 16 * m + (lane & 15);
      int K = 32 * kc + 8 * (lane >> 4) + (e < 8 ? e : e + 8);
      bwdA[xx] = (_Float16)Ws[(L * 64 + M) * 64 + K];
      continue;
    }
    xx -= 28672;
    if (xx < 2048) {
      int e = xx & 15, lane = (xx >> 4) & 31, m = xx >> 9;
      int M = 16 * m + (lane & 15);
      int K = 8 * (lane >> 4) + (e < 8 ? e : e + 8);
      float v = 0.0f;
      if (K < 3) v = W_in[K * 64 + M];
      else if (K == 3) v = b_in[M]; // bias row, multiplied by the B "ones" row
      inA[xx] = (_Float16)v;
      continue;
    }
    xx -= 2048;
    if (xx < 1024) {
      int e = xx & 15, lane = (xx >> 4) & 31, kc = xx >> 9;
      int M = lane & 15;
      int K = 32 * kc + 8 * (lane >> 4) + (e < 8 ? e : e + 8);
      outA[xx] = (_Float16)((M < 3) ? W_in[M * 64 + K] : 0.0f);
      continue;
    }
    xx -= 1024;
    if (xx < 7168) {
      int r = xx & 7, lane = (xx >> 3) & 31, m = (xx >> 8) & 3, L = xx >> 10;
      int f = 16 * m + 8 * (lane >> 4) + r;
      fwdBias[xx] = bs[L * 64 + f];
      continue;
    }
    xx -= 7168;
    {
      int r = xx & 7, lane = (xx >> 3) & 31, j = (xx >> 8) & 3, c = xx >> 10;
      int f = 16 * j + 8 * (lane >> 4) + r;
      gInit[xx] = W_out[f * 2 + c]; // d a[:,c] / d y7[f]
    }
  }
}

// ---------------- fused fwd + double-VJP curl kernel (two 16-pt tiles / wave) ----------------
// Two tiles per wave: every weight/bias load is reused by both tiles (backward
// A-tiles get 4 uses: 2 tiles x 2 cotangents), halving L2 traffic per point,
// and the doubled number of independent WMMA chains fills the WMMA->VALU
// hazard slots and load-latency gaps.
__global__ void __launch_bounds__(NW * 32)
mlp_curl_kernel(const float* __restrict__ x,
                const _Float16* __restrict__ fwdA,
                const _Float16* __restrict__ bwdA,
                const _Float16* __restrict__ inA,
                const _Float16* __restrict__ outA,
                const float* __restrict__ fwdBias,
                const float* __restrict__ gInit,
                float* __restrict__ out) {
  __shared__ uint4 stash[NW][TPW][7][4][32]; // per-wave/tile tanh stash, f16 pairs (56KB)
  const int lane = threadIdx.x & 31;
  const int wv = threadIdx.x >> 5;
  const bool hib = (lane & 16) != 0;

  const int pairIdx = blockIdx.x * NW + wv; // exactly NT/TPW wave-pairs launched
  if (pairIdx * TPW >= NT) return;          // wave-uniform guard
  const int t0 = pairIdx * TPW;

  int p[TPW];
#pragma unroll
  for (int u = 0; u < TPW; ++u) p[u] = (t0 + u) * 16 + (lane & 15);

  // ---- input linear y0 = x @ W_in + b_in (K padded to 32; bias via K==3 ones row)
  v16h B0in[TPW];
#pragma unroll
  for (int u = 0; u < TPW; ++u) {
    const float* xp = x + (size_t)p[u] * 3;
    float x0 = xp[0], x1 = xp[1], x2 = xp[2];
    u32x8 bin;
    bin[0] = hib ? 0u : pk2(x0, x1);
    bin[1] = hib ? 0u : pk2(x2, 1.0f);
#pragma unroll
    for (int v = 2; v < 8; ++v) bin[v] = 0u;
    B0in[u] = __builtin_bit_cast(v16h, bin);
  }

  v8f acc[TPW][4];
#pragma unroll
  for (int m = 0; m < 4; ++m) {
    v16h a = loadA(inA + (size_t)(m * 32 + lane) * 16);
#pragma unroll
    for (int u = 0; u < TPW; ++u)
      acc[u][m] = wmma16(a, B0in[u], vzero());
  }

  // ---- hidden layers: y_k = tanh(y_{k-1} @ W_k + b_k)
#pragma unroll
  for (int k = 1; k <= 7; ++k) {
    sched_fence();
    unsigned pkk[TPW][4][4];
#pragma unroll
    for (int u = 0; u < TPW; ++u)
#pragma unroll
      for (int j = 0; j < 4; ++j)
#pragma unroll
        for (int q = 0; q < 4; ++q)
          pkk[u][j][q] = pk2(acc[u][j][2 * q], acc[u][j][2 * q + 1]);
    if (k > 1) { // stash y_{k-1} (tanh outputs) for the backward pass
#pragma unroll
      for (int u = 0; u < TPW; ++u)
#pragma unroll
        for (int j = 0; j < 4; ++j)
          stash[wv][u][k - 2][j][lane] =
              make_uint4(pkk[u][j][0], pkk[u][j][1], pkk[u][j][2], pkk[u][j][3]);
    }
    v16h Bt[TPW][2];
#pragma unroll
    for (int u = 0; u < TPW; ++u) buildB(pkk[u], hib, Bt[u]);
#pragma unroll
    for (int m = 0; m < 4; ++m) {
      v8f bias = *(const v8f*)(fwdBias + (size_t)(((k - 1) * 4 + m) * 32 + lane) * 8);
      v16h a0 = loadA(fwdA + (size_t)((((k - 1) * 4 + m) * 2 + 0) * 32 + lane) * 16);
      v16h a1 = loadA(fwdA + (size_t)((((k - 1) * 4 + m) * 2 + 1) * 32 + lane) * 16);
#pragma unroll
      for (int u = 0; u < TPW; ++u) {
        v8f c = bias;
        c = wmma16(a0, Bt[u][0], c);
        c = wmma16(a1, Bt[u][1], c);
#pragma unroll
        for (int r = 0; r < 8; ++r) acc[u][m][r] = mytanh(c[r]);
      }
    }
  }
  // stash y7
#pragma unroll
  for (int u = 0; u < TPW; ++u)
#pragma unroll
    for (int j = 0; j < 4; ++j)
      stash[wv][u][6][j][lane] =
          make_uint4(pk2(acc[u][j][0], acc[u][j][1]), pk2(acc[u][j][2], acc[u][j][3]),
                     pk2(acc[u][j][4], acc[u][j][5]), pk2(acc[u][j][6], acc[u][j][7]));

  // ---- backward: both cotangents (e0, e1) for both tiles propagated together.
  // g state held as f16 feature-pair dwords (matches both the 16-bit WMMA D
  // layout and the B-pack format), so tanh' scaling is pure packed-f16 math.
  unsigned g0d[TPW][4][4], g1d[TPW][4][4];
  {
#pragma unroll
    for (int j = 0; j < 4; ++j) {
      v8f a = *(const v8f*)(gInit + (size_t)((0 * 4 + j) * 32 + lane) * 8);
      v8f b = *(const v8f*)(gInit + (size_t)((1 * 4 + j) * 32 + lane) * 8);
#pragma unroll
      for (int q = 0; q < 4; ++q) {
        unsigned pa = pk2(a[2 * q], a[2 * q + 1]);
        unsigned pb = pk2(b[2 * q], b[2 * q + 1]);
#pragma unroll
        for (int u = 0; u < TPW; ++u) { g0d[u][j][q] = pa; g1d[u][j][q] = pb; }
      }
    }
  }

  const h2v one2 = {(_Float16)1.0f, (_Float16)1.0f};

#pragma unroll
  for (int k = 7; k >= 1; --k) {
    sched_fence();
    v16h BA[TPW][2], BB[TPW][2];
#pragma unroll
    for (int u = 0; u < TPW; ++u) {
      unsigned gzA[4][4], gzB[4][4];
#pragma unroll
      for (int j = 0; j < 4; ++j) {
        uint4 s = stash[wv][u][k - 1][j][lane];
        unsigned sq[4] = {s.x, s.y, s.z, s.w};
#pragma unroll
        for (int q = 0; q < 4; ++q) { // g_z = g_y * (1 - y^2), packed f16
          h2v y2 = __builtin_bit_cast(h2v, sq[q]);
          h2v t2 = __builtin_elementwise_fma(y2, -y2, one2); // v_pk_fma_f16
          gzA[j][q] = __builtin_bit_cast(unsigned, (h2v)(__builtin_bit_cast(h2v, g0d[u][j][q]) * t2));
          gzB[j][q] = __builtin_bit_cast(unsigned, (h2v)(__builtin_bit_cast(h2v, g1d[u][j][q]) * t2));
        }
      }
      buildB(gzA, hib, BA[u]);
      buildB(gzB, hib, BB[u]);
    }
#pragma unroll
    for (int m = 0; m < 4; ++m) { // g_{y_{k-1}}^T = W_k @ g_z^T, 2 tiles x 2 cotangents
      v16h a0 = loadA(bwdA + (size_t)((((k - 1) * 4 + m) * 2 + 0) * 32 + lane) * 16);
      v16h a1 = loadA(bwdA + (size_t)((((k - 1) * 4 + m) * 2 + 1) * 32 + lane) * 16);
#pragma unroll
      for (int u = 0; u < TPW; ++u) {
#if HAVE_WMMA_F16D
        v8h z0 = wmma16h(a0, BA[u][0], hzero());
        v8h z1 = wmma16h(a0, BB[u][0], hzero());
        z0 = wmma16h(a1, BA[u][1], z0);
        z1 = wmma16h(a1, BB[u][1], z1);
        u32x4 u0 = __builtin_bit_cast(u32x4, z0);
        u32x4 u1 = __builtin_bit_cast(u32x4, z1);
#pragma unroll
        for (int q = 0; q < 4; ++q) { g0d[u][m][q] = u0[q]; g1d[u][m][q] = u1[q]; }
#else
        v8f z0 = wmma16(a0, BA[u][0], vzero());
        v8f z1 = wmma16(a0, BB[u][0], vzero());
        z0 = wmma16(a1, BA[u][1], z0);
        z1 = wmma16(a1, BB[u][1], z1);
#pragma unroll
        for (int q = 0; q < 4; ++q) {
          g0d[u][m][q] = pk2(z0[2 * q], z0[2 * q + 1]);
          g1d[u][m][q] = pk2(z1[2 * q], z1[2 * q + 1]);
        }
#endif
      }
    }
  }

  // dx^T = W_in @ g_{y0}^T -> D[i][point], i = input dim in vgprs r=0..2, lanes 0..15
  sched_fence();
  v16h oa0 = loadA(outA + (size_t)(0 * 32 + lane) * 16);
  v16h oa1 = loadA(outA + (size_t)(1 * 32 + lane) * 16);
#pragma unroll
  for (int u = 0; u < TPW; ++u) {
    v16h BA[2], BB[2];
    buildB(g0d[u], hib, BA);
    buildB(g1d[u], hib, BB);
    v8f d0 = wmma16(oa0, BA[0], vzero());
    v8f d1 = wmma16(oa0, BB[0], vzero());
    d0 = wmma16(oa1, BA[1], d0);
    d1 = wmma16(oa1, BB[1], d1);
    // u = [dadx1, -dadx0, dpdx0, dpdx1]
    if (!hib) {
      float4 o = make_float4(d0[1], -d0[0], d1[0], d1[1]);
      *(float4*)(out + (size_t)p[u] * 4) = o;
    }
  }
}

// ---------------- host entry ----------------
extern "C" void kernel_launch(void* const* d_in, const int* in_sizes, int n_in,
                              void* d_out, int out_size, void* d_ws, size_t ws_size,
                              hipStream_t stream) {
  (void)in_sizes; (void)n_in; (void)out_size; (void)ws_size;
  const float* x     = (const float*)d_in[0];
  const float* W_in  = (const float*)d_in[1];
  const float* b_in  = (const float*)d_in[2];
  const float* Ws    = (const float*)d_in[3];
  const float* bs    = (const float*)d_in[4];
  const float* W_out = (const float*)d_in[5];
  float* out = (float*)d_out;

  _Float16* fwdA = (_Float16*)d_ws;
  _Float16* bwdA = fwdA + 28672;
  _Float16* inA  = bwdA + 28672;
  _Float16* outA = inA + 2048;
  float* fwdBias = (float*)(outA + 1024);
  float* gInit   = fwdBias + 7168;
  // total scratch: 157,696 bytes

  prep_kernel<<<136, 512, 0, stream>>>(W_in, b_in, Ws, bs, W_out,
                                       fwdA, bwdA, inA, outA, fwdBias, gInit);
  mlp_curl_kernel<<<NT / (NW * TPW), NW * 32, 0, stream>>>(x, fwdA, bwdA, inA, outA,
                                                           fwdBias, gInit, out);
}
